// EncoderDecoderAttentionHead_25666724561361
// MI455X (gfx1250) — matrix operand order repaired and checked by reading, not compiled, not run on hardware
//
#include <hip/hip_runtime.h>
#include <hip/hip_bf16.h>

// ---------------------------------------------------------------------------
// Types for CDNA5 WMMA (wave32, 16x16x32 bf16 -> f32)
// ---------------------------------------------------------------------------
typedef __attribute__((ext_vector_type(16))) __bf16          v16bf;
typedef __attribute__((ext_vector_type(16))) unsigned short  v16u;
typedef __attribute__((ext_vector_type(8)))  float           v8f;

template <typename T> struct is_bf16raw { static constexpr bool value = false; };
template <> struct is_bf16raw<unsigned short> { static constexpr bool value = true; };

// Native fp32 -> bf16 convert (single VALU op on gfx1250, RNE).
__device__ __forceinline__ unsigned short f2bf(float f) {
  __bf16 b = (__bf16)f;
  return __builtin_bit_cast(unsigned short, b);
}

// Stage 16 contiguous source elements into LDS as bf16 (4B-aligned dest).
__device__ __forceinline__ void stage16(const float* __restrict__ g,
                                        unsigned short* __restrict__ l) {
  const float4* gf = (const float4*)g;
#pragma unroll
  for (int i = 0; i < 4; ++i) {
    float4 v = gf[i];
    unsigned p0 = (unsigned)f2bf(v.x) | ((unsigned)f2bf(v.y) << 16);
    unsigned p1 = (unsigned)f2bf(v.z) | ((unsigned)f2bf(v.w) << 16);
    ((unsigned*)l)[2 * i + 0] = p0;
    ((unsigned*)l)[2 * i + 1] = p1;
  }
}
__device__ __forceinline__ void stage16(const unsigned short* __restrict__ g,
                                        unsigned short* __restrict__ l) {
#pragma unroll
  for (int i = 0; i < 8; ++i)
    ((unsigned*)l)[i] = ((const unsigned*)g)[i];
}

// Async DMA: 32 contiguous bytes global -> LDS per thread, no VGPR data,
// tracked by ASYNCcnt (GLOBAL_LOAD_ASYNC_TO_LDS_B128; INST_OFFSET is added
// to both the LDS and global addresses per ISA 08_async_tensor 4.4).
__device__ __forceinline__ void stage16_async(const unsigned short* __restrict__ g,
                                              unsigned short* __restrict__ l) {
  unsigned lds = (unsigned)(unsigned long long)(const void*)l;  // LDS aperture: low 32b
  asm volatile("global_load_async_to_lds_b128 %0, %1, off\n\t"
               "global_load_async_to_lds_b128 %0, %1, off offset:16"
               :: "v"(lds), "v"((const void*)g)
               : "memory");
}

// Load 4 contiguous source elements as bf16 raw shorts.
__device__ __forceinline__ void load4bf(const float* __restrict__ g,
                                        unsigned short* r) {
  float4 v = *(const float4*)g;
  r[0] = f2bf(v.x); r[1] = f2bf(v.y); r[2] = f2bf(v.z); r[3] = f2bf(v.w);
}
__device__ __forceinline__ void load4bf(const unsigned short* __restrict__ g,
                                        unsigned short* r) {
  uint2 u = *(const uint2*)g;
  r[0] = (unsigned short)u.x; r[1] = (unsigned short)(u.x >> 16);
  r[2] = (unsigned short)u.y; r[3] = (unsigned short)(u.y >> 16);
}

// ---------------------------------------------------------------------------
// Tiled WMMA GEMM:  Out[M,N] = scale * (A[M,K] x B) [+ bias[N]]
//   BTRANS=1 : B row-major [N,K]  (A x B^T)
//   BTRANS=0 : B row-major [K,N]  (staged with a 4x4 register transpose)
// Block tile 128x128, 256 threads = 8 waves, wave tile 32x64, K-step 32.
// LDS double-buffered; B always stored n-major [n][k] so every fragment
// load is a pair of 16B-aligned ds_load_b128. bf16-source A (and B when BTRANS) is
// staged with GLOBAL_LOAD_ASYNC_TO_LDS_B128 (ASYNCcnt).
// ---------------------------------------------------------------------------
#define BM 128
#define BN 128
#define BK 32
#define LDA_SH 40   // row stride in ushorts (32 + 8 pad; 80B = 5*16B)
#define LDB_SH 40   // Bt row stride [n][k]

template <typename AT, typename BT, bool BTRANS, bool OUTBF, bool HAS_BIAS>
__global__ __launch_bounds__(256)
void gemm_wmma(const AT* __restrict__ A, int lda,
               const BT* __restrict__ B, int ldb,
               void* __restrict__ Out, int ldo,
               const float* __restrict__ bias,
               float scale, int K)
{
  __shared__ unsigned short sA[2][BM * LDA_SH];
  __shared__ unsigned short sBt[2][BN * LDB_SH];

  constexpr bool ASYNC_A = is_bf16raw<AT>::value;
  constexpr bool ASYNC_B = BTRANS && is_bf16raw<BT>::value;

  const int t    = threadIdx.x;
  const int lane = t & 31;
  const int wid  = t >> 5;
  const int wm   = wid >> 1;          // 0..3  (M direction)
  const int wn   = wid & 1;           // 0..1  (N direction)
  const int m0   = blockIdx.y * BM;
  const int n0   = blockIdx.x * BN;

  const int mlo = lane & 15;
  const int hi  = lane >> 4;          // half-wave select

  v8f acc[2][4];
#pragma unroll
  for (int i = 0; i < 2; ++i)
#pragma unroll
    for (int j = 0; j < 4; ++j) {
      v8f z = {0.f, 0.f, 0.f, 0.f, 0.f, 0.f, 0.f, 0.f};
      acc[i][j] = z;
    }

  // staging coordinates
  const int arow = t >> 1;            // 0..127, 16 contiguous elems per thread
  const int acol = (t & 1) << 4;      // 0 or 16
  const int k4   = (t >> 5) << 2;     // 0..28  (NN transpose staging)
  const int n4   = (t & 31) << 2;     // 0..124

  auto stage = [&](int buf, int k0) {
    // A tile [128 x 32], k-contiguous
    const AT* ga = A + (size_t)(m0 + arow) * lda + k0 + acol;
    unsigned short* la = &sA[buf][arow * LDA_SH + acol];
    if constexpr (ASYNC_A) stage16_async(ga, la);
    else                   stage16(ga, la);

    if constexpr (BTRANS) {
      // B row-major [N,K]: tile rows are k-contiguous already
      const BT* gb = B + (size_t)(n0 + arow) * ldb + k0 + acol;
      unsigned short* lb = &sBt[buf][arow * LDB_SH + acol];
      if constexpr (ASYNC_B) stage16_async(gb, lb);
      else                   stage16(gb, lb);
    } else {
      // B row-major [K,N]: 4x4 register transpose into [n][k]
      unsigned short r[4][4];
#pragma unroll
      for (int rr = 0; rr < 4; ++rr)
        load4bf(B + (size_t)(k0 + k4 + rr) * ldb + n0 + n4, r[rr]);
#pragma unroll
      for (int c = 0; c < 4; ++c) {
        uint2 w;
        w.x = (unsigned)r[0][c] | ((unsigned)r[1][c] << 16);
        w.y = (unsigned)r[2][c] | ((unsigned)r[3][c] << 16);
        *(uint2*)&sBt[buf][(n4 + c) * LDB_SH + k4] = w;
      }
    }
  };

  const int nk = K / BK;
  stage(0, 0);
  if constexpr (ASYNC_A || ASYNC_B)
    asm volatile("s_wait_asynccnt 0x0" ::: "memory");
  __syncthreads();

  for (int kt = 0; kt < nk; ++kt) {
    const int cur = kt & 1;

    // prefetch tile kt+2 into GL2 (global_prefetch_b8)
    if (kt + 2 < nk) {
      const int kp = (kt + 2) * BK;
      __builtin_prefetch(A + (size_t)(m0 + arow) * lda + kp + acol, 0, 3);
      if (BTRANS)
        __builtin_prefetch(B + (size_t)(n0 + arow) * ldb + kp + acol, 0, 3);
      else
        __builtin_prefetch(B + (size_t)(kp + k4) * ldb + n0 + n4, 0, 3);
    }

    // stage tile kt+1 into the other buffer while computing from cur
    if (kt + 1 < nk) stage(cur ^ 1, (kt + 1) * BK);

    // ---- A fragments (16x32): k = e + 16*hi ----
    v16bf afrag[2];
#pragma unroll
    for (int i = 0; i < 2; ++i) {
      const unsigned short* q =
          &sA[cur][(wm * 32 + i * 16 + mlo) * LDA_SH + hi * 16];
      v16u u;
#pragma unroll
      for (int e = 0; e < 16; ++e) u[e] = q[e];
      afrag[i] = __builtin_bit_cast(v16bf, u);
    }

    // ---- B fragments (32x16): k = e + 16*hi, from n-major LDS ----
    v16bf bfrag[4];
#pragma unroll
    for (int j = 0; j < 4; ++j) {
      const unsigned short* q =
          &sBt[cur][(wn * 64 + j * 16 + mlo) * LDB_SH + hi * 16];
      v16u u;
#pragma unroll
      for (int e = 0; e < 16; ++e) u[e] = q[e];
      bfrag[j] = __builtin_bit_cast(v16bf, u);
    }

    // ---- 8 WMMAs per wave per K-step ----
#pragma unroll
    for (int i = 0; i < 2; ++i)
#pragma unroll
      for (int j = 0; j < 4; ++j)
        acc[i][j] = __builtin_amdgcn_wmma_f32_16x16x32_bf16(
            false, afrag[i], false, bfrag[j],
            (short)0, acc[i][j], false, false);

    // close double-buffer handoff: staged DMA + stores visible after barrier
    if constexpr (ASYNC_A || ASYNC_B)
      asm volatile("s_wait_asynccnt 0x0" ::: "memory");
    __syncthreads();
  }

  // ---- epilogue: C/D layout m = r + 8*hi, n = lane&15 ----
#pragma unroll
  for (int j = 0; j < 4; ++j) {
    const int n = n0 + wn * 64 + j * 16 + mlo;
    const float bv = HAS_BIAS ? bias[n] : 0.0f;
#pragma unroll
    for (int i = 0; i < 2; ++i)
#pragma unroll
      for (int r = 0; r < 8; ++r) {
        const int m = m0 + wm * 32 + i * 16 + hi * 8 + r;
        const float v = acc[i][j][r] * scale + bv;
        if (OUTBF)
          ((unsigned short*)Out)[(size_t)m * ldo + n] = f2bf(v);
        else
          ((float*)Out)[(size_t)m * ldo + n] = v;
      }
  }
}

// ---------------------------------------------------------------------------
// Row softmax: one block per row of e[S,T]; writes bf16 probabilities.
// ---------------------------------------------------------------------------
__global__ __launch_bounds__(256)
void softmax_rows(const float* __restrict__ E,
                  unsigned short* __restrict__ Abf, int T)
{
  __shared__ float red[256];
  const int row = blockIdx.x;
  const int t   = threadIdx.x;
  const float* e = E + (size_t)row * T + t * 16;

  float v[16];
  const float4* e4 = (const float4*)e;
#pragma unroll
  for (int i = 0; i < 4; ++i) {
    float4 x = e4[i];
    v[4 * i] = x.x; v[4 * i + 1] = x.y; v[4 * i + 2] = x.z; v[4 * i + 3] = x.w;
  }

  float mx = -3.402823466e38f;
#pragma unroll
  for (int i = 0; i < 16; ++i) mx = fmaxf(mx, v[i]);
  red[t] = mx; __syncthreads();
  for (int s = 128; s > 0; s >>= 1) {
    if (t < s) red[t] = fmaxf(red[t], red[t + s]);
    __syncthreads();
  }
  mx = red[0]; __syncthreads();

  float sum = 0.f;
#pragma unroll
  for (int i = 0; i < 16; ++i) { v[i] = __expf(v[i] - mx); sum += v[i]; }
  red[t] = sum; __syncthreads();
  for (int s = 128; s > 0; s >>= 1) {
    if (t < s) red[t] += red[t + s];
    __syncthreads();
  }
  const float inv = 1.0f / red[0];

  unsigned short* o = Abf + (size_t)row * T + t * 16;
#pragma unroll
  for (int i = 0; i < 8; ++i) {
    unsigned p = (unsigned)f2bf(v[2 * i] * inv) |
                 ((unsigned)f2bf(v[2 * i + 1] * inv) << 16);
    ((unsigned*)o)[i] = p;
  }
}

// ---------------------------------------------------------------------------
// Host side
// ---------------------------------------------------------------------------
extern "C" void kernel_launch(void* const* d_in, const int* in_sizes, int n_in,
                              void* d_out, int out_size, void* d_ws, size_t ws_size,
                              hipStream_t stream)
{
  const int S = 4096, T = 4096, D = 2048;
  const float* x   = (const float*)d_in[0];   // [S, D]
  const float* enc = (const float*)d_in[1];   // [2, T, S]
  const float* Wq  = (const float*)d_in[2];   // [D, D]
  const float* bq  = (const float*)d_in[3];   // [D]
  float* z = (float*)d_out;                   // [S, D]

  // scratch carve-out (~144 MB)
  char* ws = (char*)d_ws;
  float*          e   = (float*)ws;          ws += (size_t)S * T * sizeof(float);
  unsigned short* abf = (unsigned short*)ws; ws += (size_t)S * T * 2;
  unsigned short* qbf = (unsigned short*)ws; ws += (size_t)S * D * 2;
  unsigned short* kbf = (unsigned short*)ws; ws += (size_t)T * D * 2;
  unsigned short* vbf = (unsigned short*)ws; ws += (size_t)T * D * 2;

  dim3 blk(256);

  // 1) q_bf = bf16(x @ Wq^T + bq)                [S,D]
  gemm_wmma<float, float, true, true, true>
      <<<dim3(D / BN, S / BM), blk, 0, stream>>>(x, D, Wq, D, qbf, D, bq, 1.0f, D);

  // 2) k_bf = bf16(enc0 @ x), v_bf = bf16(enc1 @ x)   [T,D]
  gemm_wmma<float, float, false, true, false>
      <<<dim3(D / BN, T / BM), blk, 0, stream>>>(enc, S, x, D, kbf, D, nullptr,
                                                 1.0f, S);
  gemm_wmma<float, float, false, true, false>
      <<<dim3(D / BN, T / BM), blk, 0, stream>>>(enc + (size_t)T * S, S, x, D,
                                                 vbf, D, nullptr, 1.0f, S);

  // 3) e = (q_bf @ k_bf^T) * rsqrt(D)           [S,T]  (fully async-staged)
  gemm_wmma<unsigned short, unsigned short, true, false, false>
      <<<dim3(T / BN, S / BM), blk, 0, stream>>>(qbf, D, kbf, D, e, T, nullptr,
                                                 0.022097086912079608f, D);

  // 4) softmax rows -> bf16 probabilities       [S,T]
  softmax_rows<<<dim3(S), blk, 0, stream>>>(e, abf, T);

  // 5) z = a @ v_bf                             [S,D]  (A async-staged)
  gemm_wmma<unsigned short, unsigned short, false, false, false>
      <<<dim3(D / BN, S / BM), blk, 0, stream>>>(abf, T, vbf, D, z, D, nullptr,
                                                 1.0f, T);
}
